// Allegro_Module_20229295964902
// MI455X (gfx1250) — matrix operand order repaired and checked by reading, not compiled, not run on hardware
//
#include <hip/hip_runtime.h>
#include <hip/hip_bf16.h>
#include <math.h>

// ---------------------------------------------------------------------------
// CDNA5 (gfx1250) Allegro-style module.
//   - MLP GEMMs via v_wmma_f32_16x16x32_f16, 4 M-tiles (64 rows) per wave so
//     each weight (B) fragment load feeds 4 WMMAs (4x less L2 weight traffic).
//   - Branch-free epilogues; 32-bit indexing -> GVS addressing w/ scale_offset.
//   - proj kernel kept under 256 VGPRs (no VGPR-MSB mode churn).
//   - global_prefetch_b8 warms later-stage weights during stage-1 compute.
//   - env_embed scatter via global_atomic_add_f32 into L2-resident atom arrays.
// Fragment layout (per CDNA5 ISA tables):
//   A 16x32 f16 : lane (r=lane&15, h=lane>>4) holds row r,
//                 v16h elems 0..7 = K[h*8 .. h*8+7], elems 8..15 = K[16+h*8 ..]
//   B 32x16 f16 : lane holds column n0+r, v16h elems 0..7 = K[h*8 .. h*8+7], elems 8..15 = K[16+h*8 ..]
//   C/D 16x16 f32: v8f elem i = D[8*h + i][n0 + r]
// ---------------------------------------------------------------------------

typedef __attribute__((ext_vector_type(16))) _Float16 v16h;
typedef __attribute__((ext_vector_type(8)))  _Float16 v8h;
typedef __attribute__((ext_vector_type(8)))  float    v8f;

#define NATOMS 5000
static constexpr float NORM_F      = 0.17677669529663687f;   // 1/sqrt(32)
static constexpr float INV_SQRT3_F = 0.57735026918962576f;

__device__ __forceinline__ float silu_f(float x) { return x / (1.0f + __expf(-x)); }

// Warm the cache hierarchy for a weight matrix (one prefetch per 128B line,
// striped across the block's threads). Emits global_prefetch_b8; no counters.
template<int BYTES>
__device__ __forceinline__ void prefetch_weights(const _Float16* W, int tid, int nthreads)
{
    constexpr int LINES = BYTES / 128;
    for (int i = tid; i < LINES; i += nthreads)
        __builtin_prefetch((const char*)W + i * 128, 0, 3);
}

// MT 16x16 output tiles (rows mt*16..mt*16+15) of A[(16*MT) x K] * W[KxN].
// A: f16 row-major (LDS or global), W: f16 row-major global.
template<int K, int MT>
__device__ __forceinline__ void wmma_tileM(v8f acc[MT],
                                           const _Float16* A, int lda,
                                           const _Float16* W, int ldw,
                                           int n0, int lane)
{
    const int r  = lane & 15;
    const int hs = lane >> 4;
#pragma unroll
    for (int kt = 0; kt < K / 32; ++kt) {
        const _Float16* wcol = W + (kt * 32 + hs * 8) * ldw + n0 + r;
        v16h b;
#pragma unroll
        for (int s = 0; s < 8; ++s) {
            b[s]     = wcol[s * ldw];
            b[s + 8] = wcol[(16 + s) * ldw];
        }
#pragma unroll
        for (int mt = 0; mt < MT; ++mt) {
            const _Float16* arow = A + (mt * 16 + r) * lda + kt * 32 + hs * 8;
            v8h alo = *(const v8h*)(arow);
            v8h ahi = *(const v8h*)(arow + 16);
            v16h a = __builtin_shufflevector(alo, ahi, 0,1,2,3,4,5,6,7,8,9,10,11,12,13,14,15);
            acc[mt] = __builtin_amdgcn_wmma_f32_16x16x32_f16(false, a, false, b, (short)0,
                                                             acc[mt], false, false);
        }
    }
}

// Copy ROWS x COLS f16 (contiguous) global -> per-wave LDS panel.
template<int COLS, int ROWS>
__device__ __forceinline__ void load_rows_to_lds(_Float16* dst, const _Float16* src, int lane)
{
    constexpr int CHUNKS = ROWS * COLS / 8;   // multiple of 32 for our shapes
#pragma unroll
    for (int i = 0; i < CHUNKS / 32; ++i) {
        int c = i * 32 + lane;
        *(v8h*)(dst + c * 8) = *(const v8h*)(src + c * 8);
    }
}

// GEMM stage with SiLU, output written back to an LDS f16 panel (64 rows).
template<int K, int N>
__device__ __forceinline__ void mlp_stage_silu(const _Float16* Abuf, int lda,
                                               const _Float16* W,
                                               _Float16* Obuf, int ldo, int lane)
{
    const int r = lane & 15, hs = lane >> 4;
#pragma unroll 2
    for (int nt = 0; nt < N / 16; ++nt) {
        v8f acc[4];
#pragma unroll
        for (int mt = 0; mt < 4; ++mt) acc[mt] = (v8f){0.f,0.f,0.f,0.f,0.f,0.f,0.f,0.f};
        wmma_tileM<K, 4>(acc, Abuf, lda, W, N, nt * 16, lane);
#pragma unroll
        for (int mt = 0; mt < 4; ++mt)
#pragma unroll
            for (int i = 0; i < 8; ++i)
                Obuf[(mt * 16 + 8 * hs + i) * ldo + nt * 16 + r] =
                    (_Float16)silu_f(acc[mt][i]);
    }
}

// ------------------------- proj = scalar_embed @ W_proj --------------------
// 4 waves/block, 64 rows/wave; A fragments live in registers (K=64 -> 64 VGPRs).
__global__ void __launch_bounds__(128) proj_kernel(
    const _Float16* __restrict__ Xse,   // [E,64]
    const _Float16* __restrict__ Wp,    // [64,256]
    float* __restrict__ out,            // [E,384]
    _Float16* __restrict__ X0,          // [E,192]
    _Float16* __restrict__ X1,          // [E,320]
    float* __restrict__ envw,           // [E,128]
    int E)
{
    const int wave = threadIdx.x >> 5, lane = threadIdx.x & 31;
    const int m0 = (blockIdx.x * 4 + wave) * 64;
    if (m0 >= E) return;
    const int r = lane & 15, hs = lane >> 4;
    const _Float16* A = Xse + m0 * 64;

    // cols 0..127 -> acc0 (out) + f16 copies into X0/X1
#pragma unroll 1
    for (int nt = 0; nt < 8; ++nt) {
        v8f acc[4];
#pragma unroll
        for (int mt = 0; mt < 4; ++mt) acc[mt] = (v8f){0.f,0.f,0.f,0.f,0.f,0.f,0.f,0.f};
        wmma_tileM<64, 4>(acc, A, 64, Wp, 256, nt * 16, lane);
        const int col = nt * 16 + r;
#pragma unroll
        for (int mt = 0; mt < 4; ++mt) {
            const int row0 = m0 + mt * 16 + 8 * hs;
#pragma unroll
            for (int i = 0; i < 8; ++i) {
                float v = acc[mt][i];
                out[(row0 + i) * 384 + col] = v;
                X0[(row0 + i) * 192 + col] = (_Float16)v;
                X1[(row0 + i) * 320 + col] = (_Float16)v;
            }
        }
    }
    // cols 128..255 -> env_w
#pragma unroll 1
    for (int nt = 8; nt < 16; ++nt) {
        v8f acc[4];
#pragma unroll
        for (int mt = 0; mt < 4; ++mt) acc[mt] = (v8f){0.f,0.f,0.f,0.f,0.f,0.f,0.f,0.f};
        wmma_tileM<64, 4>(acc, A, 64, Wp, 256, nt * 16, lane);
        const int col = nt * 16 + r - 128;
#pragma unroll
        for (int mt = 0; mt < 4; ++mt) {
            const int row0 = m0 + mt * 16 + 8 * hs;
#pragma unroll
            for (int i = 0; i < 8; ++i)
                envw[(row0 + i) * 128 + col] = acc[mt][i];
        }
    }
}

// ------------------------- MLP0: [E,192]->256->256->256 -------------------
// 2 waves/block, 64 rows/wave; per-wave LDS ping-pong activation panels.
__global__ void __launch_bounds__(64) mlp0_kernel(
    const _Float16* __restrict__ X0,    // [E,192]
    const _Float16* __restrict__ W1,    // [192,256]
    const _Float16* __restrict__ W2,    // [256,256]
    const _Float16* __restrict__ W3,    // [256,256]
    float* __restrict__ out,            // [E,384] cols 128..255 (acc1)
    _Float16* __restrict__ X1,          // [E,320] cols 128..255
    float* __restrict__ envw,           // [E,128]
    int E)
{
    extern __shared__ _Float16 lds[];
    const int wave = threadIdx.x >> 5, lane = threadIdx.x & 31;
    _Float16* bufA = lds + wave * (320 + 256) * 64;
    _Float16* bufB = bufA + 320 * 64;
    const int m0 = (blockIdx.x * 2 + wave) * 64;
    if (m0 >= E) return;
    const int r = lane & 15, hs = lane >> 4;

    // Warm later-stage weights while stage 1 runs (global_prefetch_b8).
    prefetch_weights<256 * 256 * 2>(W2, threadIdx.x, 64);
    prefetch_weights<256 * 256 * 2>(W3, threadIdx.x, 64);

    load_rows_to_lds<192, 64>(bufA, X0 + m0 * 192, lane);
    mlp_stage_silu<192, 256>(bufA, 192, W1, bufB, 256, lane);
    mlp_stage_silu<256, 256>(bufB, 256, W2, bufA, 256, lane);

    // cols 0..127 -> acc1 (out cols 128..255) + f16 copy into X1
#pragma unroll 2
    for (int nt = 0; nt < 8; ++nt) {
        v8f acc[4];
#pragma unroll
        for (int mt = 0; mt < 4; ++mt) acc[mt] = (v8f){0.f,0.f,0.f,0.f,0.f,0.f,0.f,0.f};
        wmma_tileM<256, 4>(acc, bufA, 256, W3, 256, nt * 16, lane);
        const int col = nt * 16 + r;
#pragma unroll
        for (int mt = 0; mt < 4; ++mt) {
            const int row0 = m0 + mt * 16 + 8 * hs;
#pragma unroll
            for (int i = 0; i < 8; ++i) {
                float v = acc[mt][i];
                out[(row0 + i) * 384 + 128 + col] = v;
                X1[(row0 + i) * 320 + 128 + col] = (_Float16)v;
            }
        }
    }
    // cols 128..255 -> env_w
#pragma unroll 2
    for (int nt = 8; nt < 16; ++nt) {
        v8f acc[4];
#pragma unroll
        for (int mt = 0; mt < 4; ++mt) acc[mt] = (v8f){0.f,0.f,0.f,0.f,0.f,0.f,0.f,0.f};
        wmma_tileM<256, 4>(acc, bufA, 256, W3, 256, nt * 16, lane);
        const int col = nt * 16 + r - 128;
#pragma unroll
        for (int mt = 0; mt < 4; ++mt) {
            const int row0 = m0 + mt * 16 + 8 * hs;
#pragma unroll
            for (int i = 0; i < 8; ++i)
                envw[(row0 + i) * 128 + col] = acc[mt][i];
        }
    }
}

// ------------------------- MLP1: [E,320]->256->256->128 -------------------
__global__ void __launch_bounds__(64) mlp1_kernel(
    const _Float16* __restrict__ X1,    // [E,320]
    const _Float16* __restrict__ W1,    // [320,256]
    const _Float16* __restrict__ W2,    // [256,256]
    const _Float16* __restrict__ W3,    // [256,128]
    float* __restrict__ out,            // [E,384] cols 256..383 (acc2)
    int E)
{
    extern __shared__ _Float16 lds[];
    const int wave = threadIdx.x >> 5, lane = threadIdx.x & 31;
    _Float16* bufA = lds + wave * (320 + 256) * 64;
    _Float16* bufB = bufA + 320 * 64;
    const int m0 = (blockIdx.x * 2 + wave) * 64;
    if (m0 >= E) return;
    const int r = lane & 15, hs = lane >> 4;

    prefetch_weights<256 * 256 * 2>(W2, threadIdx.x, 64);
    prefetch_weights<256 * 128 * 2>(W3, threadIdx.x, 64);

    load_rows_to_lds<320, 64>(bufA, X1 + m0 * 320, lane);
    mlp_stage_silu<320, 256>(bufA, 320, W1, bufB, 256, lane);
    mlp_stage_silu<256, 256>(bufB, 256, W2, bufA, 256, lane);
#pragma unroll 2
    for (int nt = 0; nt < 8; ++nt) {
        v8f acc[4];
#pragma unroll
        for (int mt = 0; mt < 4; ++mt) acc[mt] = (v8f){0.f,0.f,0.f,0.f,0.f,0.f,0.f,0.f};
        wmma_tileM<256, 4>(acc, bufA, 256, W3, 128, nt * 16, lane);
        const int col = nt * 16 + r;
#pragma unroll
        for (int mt = 0; mt < 4; ++mt) {
            const int row0 = m0 + mt * 16 + 8 * hs;
#pragma unroll
            for (int i = 0; i < 8; ++i)
                out[(row0 + i) * 384 + 256 + col] = acc[mt][i];
        }
    }
}

// ------------------------- env_embed scatter -------------------------------
__global__ void env_scatter_kernel(const int* __restrict__ center,
                                   const float* __restrict__ tb,     // [E,4]
                                   const float* __restrict__ envw,   // [E,128]
                                   float* __restrict__ sa,           // [A,64]
                                   float* __restrict__ va,           // [A,64,3]
                                   int E)
{
    int idx = blockIdx.x * blockDim.x + threadIdx.x;
    if (idx >= E * 64) return;
    int e = idx >> 6, c = idx & 63;
    int a = center[e];
    float w0 = envw[e * 128 + c];
    float w1 = envw[e * 128 + 64 + c];
    const float* t = tb + e * 4;
    atomicAdd(&sa[a * 64 + c], NORM_F * t[0] * w0);
    float s = NORM_F * w1;
    atomicAdd(&va[a * 192 + c * 3 + 0], s * t[1]);
    atomicAdd(&va[a * 192 + c * 3 + 1], s * t[2]);
    atomicAdd(&va[a * 192 + c * 3 + 2], s * t[3]);
}

// ------------------------- tensor product 0 (o0, ov) -----------------------
__global__ void tp0_kernel(const int* __restrict__ center,
                           const float* __restrict__ tf,    // [E,64,4]
                           const float* __restrict__ sa,
                           const float* __restrict__ va,
                           const float* __restrict__ wtp0,  // [4,64]
                           float* __restrict__ o0,          // [E,64]
                           float* __restrict__ ov,          // [E,64,3]
                           _Float16* __restrict__ X0,       // [E,192] cols 128..191
                           int E)
{
    int idx = blockIdx.x * blockDim.x + threadIdx.x;
    if (idx >= E * 64) return;
    int e = idx >> 6, c = idx & 63;
    int a = center[e];
    float es  = sa[a * 64 + c];
    float ev0 = va[a * 192 + c * 3 + 0];
    float ev1 = va[a * 192 + c * 3 + 1];
    float ev2 = va[a * 192 + c * 3 + 2];
    const float* tfe = tf + e * 256 + c * 4;
    float t0 = tfe[0], tv0 = tfe[1], tv1 = tfe[2], tv2 = tfe[3];
    float dot = ev0 * tv0 + ev1 * tv1 + ev2 * tv2;
    float o = wtp0[c] * es * t0 + wtp0[64 + c] * INV_SQRT3_F * dot;
    o0[idx] = o;
    X0[e * 192 + 128 + c] = (_Float16)o;
    float ka = wtp0[128 + c] * es, kb = wtp0[192 + c] * t0;
    ov[e * 192 + c * 3 + 0] = ka * tv0 + kb * ev0;
    ov[e * 192 + c * 3 + 1] = ka * tv1 + kb * ev1;
    ov[e * 192 + c * 3 + 2] = ka * tv2 + kb * ev2;
}

// ------------------------- tensor product 1 (o_last) -----------------------
__global__ void tp1_kernel(const int* __restrict__ center,
                           const float* __restrict__ sa,
                           const float* __restrict__ va,
                           const float* __restrict__ wtp1,  // [2,64]
                           const float* __restrict__ o0,    // [E,64]
                           const float* __restrict__ ov,    // [E,64,3]
                           _Float16* __restrict__ X1,       // [E,320] cols 256..319
                           int E)
{
    int idx = blockIdx.x * blockDim.x + threadIdx.x;
    if (idx >= E * 64) return;
    int e = idx >> 6, c = idx & 63;
    int a = center[e];
    float es  = sa[a * 64 + c];
    float ev0 = va[a * 192 + c * 3 + 0];
    float ev1 = va[a * 192 + c * 3 + 1];
    float ev2 = va[a * 192 + c * 3 + 2];
    const float* ove = ov + e * 192 + c * 3;
    float dot = ev0 * ove[0] + ev1 * ove[1] + ev2 * ove[2];
    float ol = wtp1[c] * es * o0[idx] + wtp1[64 + c] * INV_SQRT3_F * dot;
    X1[e * 320 + 256 + c] = (_Float16)ol;
}

// ------------------------- utility kernels ---------------------------------
__global__ void f32_to_f16_kernel(const float* __restrict__ s, _Float16* __restrict__ d, int n)
{
    int i = blockIdx.x * blockDim.x + threadIdx.x;
    if (i < n) d[i] = (_Float16)s[i];
}

__global__ void zero_kernel(float* __restrict__ p, int n)
{
    int i = blockIdx.x * blockDim.x + threadIdx.x;
    if (i < n) p[i] = 0.0f;
}

// ---------------------------------------------------------------------------
extern "C" void kernel_launch(void* const* d_in, const int* in_sizes, int n_in,
                              void* d_out, int out_size, void* d_ws, size_t ws_size,
                              hipStream_t stream)
{
    (void)n_in; (void)out_size; (void)ws_size;
    const int*   center = (const int*)d_in[0];          // edge_index[0] = first E ints
    const float* tb     = (const float*)d_in[2];
    const float* tf     = (const float*)d_in[3];
    const float* se     = (const float*)d_in[4];
    const float* Wprojf = (const float*)d_in[5];
    const float* wtp0   = (const float*)d_in[6];
    const float* wtp1   = (const float*)d_in[7];
    const float* l0W1f  = (const float*)d_in[8];
    const float* l0W2f  = (const float*)d_in[9];
    const float* l0W3f  = (const float*)d_in[10];
    const float* l1W1f  = (const float*)d_in[11];
    const float* l1W2f  = (const float*)d_in[12];
    const float* l1W3f  = (const float*)d_in[13];
    const int E = in_sizes[2] / 4;                      // tensor_basis is [E,4]
    float* out = (float*)d_out;

    // ----- workspace carve-out -----
    char* ws = (char*)d_ws;
    size_t off = 0;
    auto carve = [&](size_t bytes) -> char* {
        char* p = ws + off;
        off += (bytes + 255) & ~(size_t)255;
        return p;
    };
    _Float16* Wp   = (_Float16*)carve((size_t)64 * 256 * 2);
    _Float16* L0W1 = (_Float16*)carve((size_t)192 * 256 * 2);
    _Float16* L0W2 = (_Float16*)carve((size_t)256 * 256 * 2);
    _Float16* L0W3 = (_Float16*)carve((size_t)256 * 256 * 2);
    _Float16* L1W1 = (_Float16*)carve((size_t)320 * 256 * 2);
    _Float16* L1W2 = (_Float16*)carve((size_t)256 * 256 * 2);
    _Float16* L1W3 = (_Float16*)carve((size_t)256 * 128 * 2);
    _Float16* Xse  = (_Float16*)carve((size_t)E * 64 * 2);
    _Float16* X0   = (_Float16*)carve((size_t)E * 192 * 2);
    _Float16* X1   = (_Float16*)carve((size_t)E * 320 * 2);
    float*    envw = (float*)carve((size_t)E * 128 * 4);
    float*    o0b  = (float*)carve((size_t)E * 64 * 4);
    float*    ovb  = (float*)carve((size_t)E * 192 * 4);
    float*    sav  = (float*)carve((size_t)NATOMS * 256 * 4);   // sa [A,64] ++ va [A,192]
    float*    sa   = sav;
    float*    va   = sav + (size_t)NATOMS * 64;

    const int T = 256;
    auto cvt = [&](const float* s, _Float16* d, int n) {
        f32_to_f16_kernel<<<(n + T - 1) / T, T, 0, stream>>>(s, d, n);
    };
    // weight + activation f16 conversion
    cvt(Wprojf, Wp,   64 * 256);
    cvt(l0W1f,  L0W1, 192 * 256);
    cvt(l0W2f,  L0W2, 256 * 256);
    cvt(l0W3f,  L0W3, 256 * 256);
    cvt(l1W1f,  L1W1, 320 * 256);
    cvt(l1W2f,  L1W2, 256 * 256);
    cvt(l1W3f,  L1W3, 256 * 128);
    cvt(se,     Xse,  E * 64);

    const int projBlocks = (E + 255) / 256;              // 4 waves x 64 rows / block
    const int mlpBlocks  = (E + 127) / 128;              // 2 waves x 64 rows / block
    const size_t ldsBytes = (size_t)2 * (320 + 256) * 64 * sizeof(_Float16); // 147456 B
    const int ecBlocks = (E * 64 + T - 1) / T;

    // proj -> acc0 / env_w
    proj_kernel<<<projBlocks, 128, 0, stream>>>(Xse, Wp, out, X0, X1, envw, E);

    // env pass 1
    zero_kernel<<<(NATOMS * 256 + T - 1) / T, T, 0, stream>>>(sav, NATOMS * 256);
    env_scatter_kernel<<<ecBlocks, T, 0, stream>>>(center, tb, envw, sa, va, E);
    tp0_kernel<<<ecBlocks, T, 0, stream>>>(center, tf, sa, va, wtp0, o0b, ovb, X0, E);

    // MLP0 -> acc1 / env_w (overwrites envw)
    mlp0_kernel<<<mlpBlocks, 64, ldsBytes, stream>>>(X0, L0W1, L0W2, L0W3, out, X1, envw, E);

    // env pass 2
    zero_kernel<<<(NATOMS * 256 + T - 1) / T, T, 0, stream>>>(sav, NATOMS * 256);
    env_scatter_kernel<<<ecBlocks, T, 0, stream>>>(center, tb, envw, sa, va, E);
    tp1_kernel<<<ecBlocks, T, 0, stream>>>(center, sa, va, wtp1, o0b, ovb, X1, E);

    // MLP1 -> acc2
    mlp1_kernel<<<mlpBlocks, 64, ldsBytes, stream>>>(X1, L1W1, L1W2, L1W3, out, E);
}
